// GatbertSelfAttention_69827578298375
// MI455X (gfx1250) — matrix operand order repaired and checked by reading, not compiled, not run on hardware
//
#include <hip/hip_runtime.h>
#include <math.h>

typedef unsigned short ushort_t;
typedef __attribute__((ext_vector_type(16))) __bf16 v16bf;
typedef __attribute__((ext_vector_type(8)))  float  v8f;

#define GB     2
#define GN     4096
#define HID    768
#define HEADS  12
#define DH     64
#define NEDGE  131072
#define MROWS  8192      // GB*GN
#define NQKV   2304      // 3*HID
#define MAXDEG 512

__device__ __forceinline__ ushort_t f2bf(float x) {
  unsigned u = __float_as_uint(x);
  unsigned r = u + 0x7fffu + ((u >> 16) & 1u);   // round-to-nearest-even
  return (ushort_t)(r >> 16);
}

union FragBF { v16bf v; float4 f4[2]; };

// ---------------- prep: fp32->bf16 A, transposed+concatenated W, bias, zero counts
__global__ void prep(const float* __restrict__ ns,
                     const float* __restrict__ Wq, const float* __restrict__ Wk,
                     const float* __restrict__ Wv,
                     const float* __restrict__ bq, const float* __restrict__ bk,
                     const float* __restrict__ bv,
                     ushort_t* __restrict__ Abf, ushort_t* __restrict__ Wtbf,
                     float* __restrict__ biasC, unsigned* __restrict__ cnt) {
  const size_t stride = (size_t)gridDim.x * blockDim.x;
  const size_t t0 = (size_t)blockIdx.x * blockDim.x + threadIdx.x;
  for (size_t i = t0; i < (size_t)MROWS * HID; i += stride) Abf[i] = f2bf(ns[i]);
  for (size_t i = t0; i < (size_t)NQKV * HID; i += stride) {
    int n = (int)(i / HID), k = (int)(i % HID);
    float w;
    if (n < HID)            w = Wq[(size_t)k * HID + n];
    else if (n < 2 * HID)   w = Wk[(size_t)k * HID + (n - HID)];
    else                    w = Wv[(size_t)k * HID + (n - 2 * HID)];
    Wtbf[i] = f2bf(w);
  }
  for (size_t i = t0; i < NQKV; i += stride)
    biasC[i] = (i < HID) ? bq[i] : (i < 2 * HID ? bk[i - HID] : bv[i - 2 * HID]);
  for (size_t i = t0; i < MROWS; i += stride) cnt[i] = 0u;
}

// ---------------- fused QKV GEMM: QKV[8192 x 2304] = A[8192x768] * Wt^T + bias
// 8 waves/block; wave tile 32x32 (2x2 WMMA 16x16x32 bf16); block tile 64x128
__global__ __launch_bounds__(256) void gemm_qkv(
    const ushort_t* __restrict__ A, const ushort_t* __restrict__ Wt,
    const float* __restrict__ biasC, float* __restrict__ QKV) {
  const int lane  = threadIdx.x & 31;
  const int wave  = threadIdx.x >> 5;
  const int m0 = blockIdx.x * 64  + (wave & 1) * 32;
  const int n0 = blockIdx.y * 128 + (wave >> 1) * 32;
  const int r  = lane & 15;
  const int hi = lane >> 4;

  v8f acc[2][2] = {};
  for (int k = 0; k < HID; k += 32) {
    FragBF a[2], b[2];
#pragma unroll
    for (int mi = 0; mi < 2; ++mi) {
      // A layout: lane&15 = row, K chunks [kb,kb+8) and [kb+16,kb+24), kb=(lane>>4)*8
      const ushort_t* p = A + (size_t)(m0 + mi * 16 + r) * HID + k + hi * 8;
      a[mi].f4[0] = *(const float4*)p;
      a[mi].f4[1] = *(const float4*)(p + 16);
    }
#pragma unroll
    for (int ni = 0; ni < 2; ++ni) {
      // B layout: lane&15 = col, K chunks [kb,kb+8) and [kb+16,kb+24), kb=(lane>>4)*8
      const ushort_t* p = Wt + (size_t)(n0 + ni * 16 + r) * HID + k + hi * 8;
      b[ni].f4[0] = *(const float4*)p;
      b[ni].f4[1] = *(const float4*)(p + 16);
    }
#pragma unroll
    for (int mi = 0; mi < 2; ++mi)
#pragma unroll
      for (int ni = 0; ni < 2; ++ni)
        acc[mi][ni] = __builtin_amdgcn_wmma_f32_16x16x32_bf16(
            false, a[mi].v, false, b[ni].v, (short)0, acc[mi][ni], false, false);
  }
#pragma unroll
  for (int mi = 0; mi < 2; ++mi) {
    const int rowBase = m0 + mi * 16 + hi * 8;   // VGPR t -> row rowBase+t
#pragma unroll
    for (int ni = 0; ni < 2; ++ni) {
      const int col = n0 + ni * 16 + r;
      const float bc = biasC[col];
#pragma unroll
      for (int t = 0; t < 8; ++t)
        QKV[(size_t)(rowBase + t) * NQKV + col] = acc[mi][ni][t] + bc;
    }
  }
}

// ---------------- CSR build over segments seg = b*GN + head_idx
__global__ void count_edges(const int* __restrict__ ei, unsigned* __restrict__ cnt) {
  int e = blockIdx.x * blockDim.x + threadIdx.x;
  if (e >= NEDGE) return;
  atomicAdd(&cnt[ei[e] * GN + ei[NEDGE + e]], 1u);
}

__global__ __launch_bounds__(1024) void scan_seg(const unsigned* __restrict__ cnt,
                                                 unsigned* __restrict__ off,
                                                 unsigned* __restrict__ cur) {
  __shared__ unsigned part[1024];
  const int tid = threadIdx.x;
  unsigned s = 0;
#pragma unroll
  for (int j = 0; j < 8; ++j) s += cnt[tid * 8 + j];
  part[tid] = s;
  __syncthreads();
  for (int d = 1; d < 1024; d <<= 1) {
    unsigned v = (tid >= d) ? part[tid - d] : 0u;
    __syncthreads();
    part[tid] += v;
    __syncthreads();
  }
  unsigned run = (tid == 0) ? 0u : part[tid - 1];
#pragma unroll
  for (int j = 0; j < 8; ++j) {
    int sIdx = tid * 8 + j;
    off[sIdx] = run;
    cur[sIdx] = run;
    run += cnt[sIdx];
  }
  if (tid == 1023) off[MROWS] = run;
}

__global__ void fill_edges(const int* __restrict__ ei, unsigned* __restrict__ cur,
                           int* __restrict__ tailS) {
  int e = blockIdx.x * blockDim.x + threadIdx.x;
  if (e >= NEDGE) return;
  unsigned pos = atomicAdd(&cur[ei[e] * GN + ei[NEDGE + e]], 1u);
  if (pos < NEDGE) tailS[pos] = ei[2 * NEDGE + e];
}

// ---------------- per-segment attention: scores, softmax, gather of V (no atomics)
__global__ __launch_bounds__(256) void seg_attn(const float* __restrict__ QKV,
                                                const unsigned* __restrict__ off,
                                                const int* __restrict__ tailS,
                                                float* __restrict__ out) {
  __shared__ __align__(16) float qrow[HID];
  __shared__ float sc[MAXDEG * HEADS];
  __shared__ int   tails[MAXDEG];
  __shared__ float hmax[HEADS], hsum[HEADS];

  const int s   = blockIdx.x;
  const int tid = threadIdx.x;
  const int base = (int)off[s];
  int deg = (int)(off[s + 1] - off[s]);
  if (deg > MAXDEG) deg = MAXDEG;
  const int rowB = s & ~(GN - 1);              // b*GN (GN power of 2)

  for (int i = tid; i < HID; i += 256) qrow[i] = QKV[(size_t)s * NQKV + i];
  for (int i = tid; i < deg; i += 256) tails[i] = tailS[base + i];
  __syncthreads();

  // per-(edge,head) scores: Q (shared, LDS) dot K (L2-resident gather)
  for (int idx = tid; idx < deg * HEADS; idx += 256) {
    int e = idx / HEADS, h = idx - e * HEADS;
    const float4* q4 = (const float4*)(qrow + h * DH);
    const float4* k4 = (const float4*)(QKV + (size_t)(rowB + tails[e]) * NQKV + HID + h * DH);
    float acc = 0.f;
#pragma unroll
    for (int d = 0; d < DH / 4; ++d) {
      float4 a = q4[d], b = k4[d];
      acc += a.x * b.x + a.y * b.y + a.z * b.z + a.w * b.w;
    }
    sc[idx] = acc * 0.125f;                    // 1/sqrt(64)
  }
  __syncthreads();

  if (tid < HEADS) {
    float m = -3.4e38f;
    for (int i = 0; i < deg; ++i) m = fmaxf(m, sc[i * HEADS + tid]);
    float su = 0.f;
    for (int i = 0; i < deg; ++i) su += expf(sc[i * HEADS + tid] - m);
    hmax[tid] = m;
    hsum[tid] = (su > 0.f) ? su : 1.f;
  }
  __syncthreads();

  for (int idx = tid; idx < deg * HEADS; idx += 256) {
    int h = idx % HEADS;
    sc[idx] = expf(sc[idx] - hmax[h]) / hsum[h];
  }
  __syncthreads();

  // out[s, o] = sum_e p[e,h] * V[tail_e, o]; coalesced V row reads across lanes
  for (int o = tid; o < HID; o += 256) {
    int h = o >> 6;
    float acc = 0.f;
    const float* vcol = QKV + 2 * HID + o;
    for (int i = 0; i < deg; ++i)
      acc += sc[i * HEADS + h] * vcol[(size_t)(rowB + tails[i]) * NQKV];
    out[(size_t)s * HID + o] = acc;
  }
}

extern "C" void kernel_launch(void* const* d_in, const int* in_sizes, int n_in,
                              void* d_out, int out_size, void* d_ws, size_t ws_size,
                              hipStream_t stream) {
  const float* ns = (const float*)d_in[0];
  const int*   ei = (const int*)d_in[1];
  // d_in[2] node_type_ids: unused by reference
  const float* Wq = (const float*)d_in[3];
  const float* bq = (const float*)d_in[4];
  const float* Wk = (const float*)d_in[5];
  const float* bk = (const float*)d_in[6];
  const float* Wv = (const float*)d_in[7];
  const float* bv = (const float*)d_in[8];
  float* out = (float*)d_out;

  char* w = (char*)d_ws;
  auto alloc = [&](size_t bytes) {
    char* p = w;
    w += (bytes + 255) & ~(size_t)255;
    return p;
  };
  ushort_t* Abf   = (ushort_t*)alloc((size_t)MROWS * HID * 2);
  ushort_t* Wtbf  = (ushort_t*)alloc((size_t)NQKV * HID * 2);
  float*    biasC = (float*)   alloc((size_t)NQKV * 4);
  float*    QKV   = (float*)   alloc((size_t)MROWS * NQKV * 4);
  unsigned* cnt   = (unsigned*)alloc((size_t)MROWS * 4);
  unsigned* off   = (unsigned*)alloc((size_t)(MROWS + 1) * 4);
  unsigned* cur   = (unsigned*)alloc((size_t)MROWS * 4);
  int*      tailS = (int*)     alloc((size_t)NEDGE * 4);
  (void)ws_size; (void)in_sizes; (void)n_in; (void)out_size;

  prep<<<dim3(1024), dim3(256), 0, stream>>>(ns, Wq, Wk, Wv, bq, bk, bv,
                                             Abf, Wtbf, biasC, cnt);
  gemm_qkv<<<dim3(MROWS / 64, NQKV / 128), dim3(256), 0, stream>>>(Abf, Wtbf, biasC, QKV);
  count_edges<<<dim3(NEDGE / 256), dim3(256), 0, stream>>>(ei, cnt);
  scan_seg<<<dim3(1), dim3(1024), 0, stream>>>(cnt, off, cur);
  fill_edges<<<dim3(NEDGE / 256), dim3(256), 0, stream>>>(ei, cur, tailS);
  seg_attn<<<dim3(MROWS), dim3(256), 0, stream>>>(QKV, off, tailS, out);
}
